// Model_43671227465945
// MI455X (gfx1250) — matrix operand order repaired and checked by reading, not compiled, not run on hardware
//
// CDNA5 (gfx1250) bidirectional-Mamba forecaster.
// GEMMs via v_wmma_f32_16x16x32_bf16 with pre-converted bf16 operands:
// inner loop = pure b128 loads + wmma (no per-element conversion/predication).
#include <hip/hip_runtime.h>

#define ACT_NONE 0
#define ACT_RELU 1
#define ACT_SOFTPLUS 2

#define BATCH 8
#define LTOK 1031              // NVAR + TFEAT
#define MROWS (BATCH * LTOK)   // 8248 real tokens
#define MPAD 8256              // 516 * 16
#define SEQ 512
#define DM 128
#define DSTATE 16
#define DTR 8
#define NVAR 1024
#define PRED 96
#define DBC_LD 48              // padded 40 -> 48

typedef __attribute__((ext_vector_type(16))) __bf16    v16bf;
typedef __attribute__((ext_vector_type(8)))  float     v8f;
typedef __attribute__((ext_vector_type(8)))  unsigned  v8u;
typedef __attribute__((ext_vector_type(4)))  unsigned  v4u;
typedef unsigned short u16;

__device__ __forceinline__ u16 f2bfu(float f) {   // round-to-nearest-even
  unsigned u = __builtin_bit_cast(unsigned, f);
  return (u16)((u + 0x7FFFu + ((u >> 16) & 1u)) >> 16);
}

// ---------------------------------------------------------------------------
// C[MPAD x N] (+)= act( Xb[MPAD x K] * Wb[N x K]^T + bias ), bf16 in, f32 acc.
// One wave per 16 x (16*NT) tile; K step 32; fragments are contiguous b128
// loads per ISA 7.12.2 wave32 layouts.  No bounds checks (shapes padded).
// ---------------------------------------------------------------------------
template <int NT, int ACT, int ACC>
__global__ __launch_bounds__(32)
void gemm_wmma(const u16* __restrict__ Xb, int lda,   // row stride (elems)
               const u16* __restrict__ Wb,            // ldw = K
               const float* __restrict__ bias,        // nullable
               float* __restrict__ C, int ldc, int K)
{
  const int tile_m  = blockIdx.x * 16;
  const int tile_n0 = blockIdx.y * 16 * NT;
  const int lane = threadIdx.x;
  const int half = lane >> 4;
  const int lidx = lane & 15;

  const size_t arow = (size_t)(tile_m + lidx) * lda + half * 8;
  size_t brow[NT];
#pragma unroll
  for (int t = 0; t < NT; ++t)
    brow[t] = (size_t)(tile_n0 + t * 16 + lidx) * K + half * 8;

  v8f acc[NT];
#pragma unroll
  for (int t = 0; t < NT; ++t) acc[t] = (v8f){};

  for (int kk = 0; kk < K; kk += 32) {
    v4u alo = *(const v4u*)(Xb + arow + kk);        // K = half*8 + 0..7
    v4u ahi = *(const v4u*)(Xb + arow + kk + 16);   // K = 16 + half*8 + 0..7
    v8u au;
#pragma unroll
    for (int j = 0; j < 4; ++j) { au[j] = alo[j]; au[4 + j] = ahi[j]; }
    v16bf a = __builtin_bit_cast(v16bf, au);
#pragma unroll
    for (int t = 0; t < NT; ++t) {
      v4u blo = *(const v4u*)(Wb + brow[t] + kk);      // K = half*8 + 0..7
      v4u bhi = *(const v4u*)(Wb + brow[t] + kk + 16); // K = 16 + half*8 + 0..7
      v8u bu;
#pragma unroll
      for (int j = 0; j < 4; ++j) { bu[j] = blo[j]; bu[4 + j] = bhi[j]; }
      v16bf b = __builtin_bit_cast(v16bf, bu);
      acc[t] = __builtin_amdgcn_wmma_f32_16x16x32_bf16(
          false, a, false, b, (short)0, acc[t], false, false);
    }
  }

#pragma unroll
  for (int t = 0; t < NT; ++t) {
    const int col = tile_n0 + t * 16 + lidx;
    const float bv = bias ? bias[col] : 0.f;
#pragma unroll
    for (int r = 0; r < 8; ++r) {
      const int row = tile_m + half * 8 + r;
      float v = acc[t][r] + bv;
      if (ACC) v += C[(size_t)row * ldc + col];
      if (ACT == ACT_RELU)     v = v > 0.f ? v : 0.f;
      if (ACT == ACT_SOFTPLUS) v = (v > 20.f) ? v : log1pf(__expf(v));
      C[(size_t)row * ldc + col] = v;
    }
  }
}

// Generic f32 -> bf16 conversion pass.
__global__ void to_bf16(const float* __restrict__ src, u16* __restrict__ dst,
                        int n)
{
  for (int i = blockIdx.x * blockDim.x + threadIdx.x; i < n;
       i += gridDim.x * blockDim.x)
    dst[i] = f2bfu(src[i]);
}

// tokens[(b*1031+t)*512+s] (bf16) from x_enc (B,SEQ,NVAR) / x_mark_enc.
__global__ void prep_tokens(const float* __restrict__ xe,
                            const float* __restrict__ xm,
                            u16* __restrict__ tok, int total)
{
  for (int i = blockIdx.x * blockDim.x + threadIdx.x; i < total;
       i += gridDim.x * blockDim.x) {
    int s = i & (SEQ - 1);
    int bt = i >> 9;
    float v = 0.f;
    if (bt < MROWS) {
      int t = bt % LTOK;
      int b = bt / LTOK;
      if (t < NVAR) {
        v = xe[((size_t)b * SEQ + s) * NVAR + t];
        if (v == -9999.0f) v = -1.0f;
      } else {
        v = xm[((size_t)b * SEQ + s) * 7 + (t - NVAR)];
      }
    }
    tok[i] = f2bfu(v);
  }
}

// Zero-pad + convert ragged weights:
//   xpw_b[idx][48][128] (rows 40..47 zero), dpw_b[idx][128][32] (cols 8..31 0)
__global__ void pack_weights(const float* __restrict__ xpw,
                             const float* __restrict__ dpw,
                             u16* __restrict__ xpw_b, u16* __restrict__ dpw_b)
{
  const int XP = 4 * 48 * DM;
  const int DP = 4 * DM * 32;
  for (int i = blockIdx.x * blockDim.x + threadIdx.x; i < XP + DP;
       i += gridDim.x * blockDim.x) {
    if (i < XP) {
      int k = i & (DM - 1);
      int n = (i >> 7) % 48;
      int idx = (i >> 7) / 48;
      xpw_b[i] = f2bfu((n < 40) ? xpw[((size_t)idx * 40 + n) * DM + k] : 0.f);
    } else {
      int j = i - XP;
      int k = j & 31;
      int n = (j >> 5) & (DM - 1);
      int idx = j >> 12;
      dpw_b[j] = f2bfu((k < DTR) ? dpw[((size_t)idx * DM + n) * DTR + k] : 0.f);
    }
  }
}

// dt columns of dbc -> zero-padded (MPAD x 32) bf16 A-matrix for dt_proj.
__global__ void pack_dt(const float* __restrict__ dbc, u16* __restrict__ dtp,
                        int total)
{
  for (int i = blockIdx.x * blockDim.x + threadIdx.x; i < total;
       i += gridDim.x * blockDim.x) {
    int k = i & 31;
    int row = i >> 5;
    float v = (k < DTR && row < MROWS) ? dbc[(size_t)row * DBC_LD + k] : 0.f;
    dtp[i] = f2bfu(v);
  }
}

// depthwise conv (k=2) along token axis + SiLU; dir=1 = mirrored conv.
// Writes f32 (for the scan) and bf16 (for the x_proj GEMM).
__global__ void conv_silu(const float* __restrict__ xz,
                          const float* __restrict__ cw,   // (DM,2)
                          const float* __restrict__ cb,   // (DM)
                          float* __restrict__ xc, u16* __restrict__ xcb16,
                          int dir, int total)
{
  for (int i = blockIdx.x * blockDim.x + threadIdx.x; i < total;
       i += gridDim.x * blockDim.x) {
    int d = i & (DM - 1);
    int row = i >> 7;
    int t = row % LTOK;
    float x0 = xz[(size_t)row * 256 + d];
    float nb;
    if (dir == 0) nb = (t > 0)        ? xz[((size_t)row - 1) * 256 + d] : 0.f;
    else          nb = (t < LTOK - 1) ? xz[((size_t)row + 1) * 256 + d] : 0.f;
    float v = nb * cw[d * 2 + 0] + x0 * cw[d * 2 + 1] + cb[d];
    float o = v / (1.f + __expf(-v));
    xc[i] = o;
    xcb16[i] = f2bfu(o);
  }
}

// Selective scan: one thread per (b,d); 16 states in registers, sequential
// over L=1031 (descending for dir=1).  Fuses y = (scan + D*xc) * silu(z);
// emits bf16 directly (y only feeds the out_proj GEMM).
__global__ __launch_bounds__(128)
void mamba_scan(const float* __restrict__ dbc,    // (M,48): dt|B|C|pad
                const float* __restrict__ delta,  // (M,128)
                const float* __restrict__ xc,     // (M,128)
                const float* __restrict__ xz,     // (M,256), z = cols 128..255
                const float* __restrict__ A_log,  // (128,16)
                const float* __restrict__ Dp,     // (128)
                u16* __restrict__ y, int dir)
{
  const int b = blockIdx.x;
  const int d = threadIdx.x;
  float A[DSTATE], h[DSTATE];
#pragma unroll
  for (int n = 0; n < DSTATE; ++n) {
    A[n] = -__expf(A_log[d * DSTATE + n]);
    h[n] = 0.f;
  }
  const float Dd = Dp[d];
  for (int step = 0; step < LTOK; ++step) {
    int t = dir ? (LTOK - 1 - step) : step;
    size_t row = (size_t)b * LTOK + t;
    float dl = delta[row * DM + d];
    float xv = xc[row * DM + d];
    const float4* B4 = (const float4*)(dbc + row * DBC_LD + DTR);
    const float4* C4 = (const float4*)(dbc + row * DBC_LD + DTR + DSTATE);
    float Bv[DSTATE], Cv[DSTATE];
#pragma unroll
    for (int q = 0; q < 4; ++q) {
      float4 bq = B4[q], cq = C4[q];
      Bv[4 * q] = bq.x; Bv[4 * q + 1] = bq.y; Bv[4 * q + 2] = bq.z; Bv[4 * q + 3] = bq.w;
      Cv[4 * q] = cq.x; Cv[4 * q + 1] = cq.y; Cv[4 * q + 2] = cq.z; Cv[4 * q + 3] = cq.w;
    }
    float yv = 0.f;
#pragma unroll
    for (int n = 0; n < DSTATE; ++n) {
      float dA = __expf(dl * A[n]);
      h[n] = dA * h[n] + dl * Bv[n] * xv;
      yv += h[n] * Cv[n];
    }
    yv += Dd * xv;
    float z = xz[row * 256 + DM + d];
    y[row * DM + d] = f2bfu(yv * (z / (1.f + __expf(-z))));
  }
}

// LayerNorm over DM=128; optional second input (residual) and bf16 mirror.
__global__ __launch_bounds__(128)
void layernorm_k(float* __restrict__ out, u16* __restrict__ outb,
                 const float* __restrict__ in0, const float* __restrict__ in1,
                 const float* __restrict__ w, const float* __restrict__ b)
{
  __shared__ float red[DM];
  const int row = blockIdx.x;
  const int d = threadIdx.x;
  float x = in0[(size_t)row * DM + d];
  if (in1) x += in1[(size_t)row * DM + d];
  red[d] = x;
  __syncthreads();
  for (int s = 64; s > 0; s >>= 1) { if (d < s) red[d] += red[d + s]; __syncthreads(); }
  float mu = red[0] * (1.f / DM);
  __syncthreads();
  float c = x - mu;
  red[d] = c * c;
  __syncthreads();
  for (int s = 64; s > 0; s >>= 1) { if (d < s) red[d] += red[d + s]; __syncthreads(); }
  float var = red[0] * (1.f / DM);
  float o = c * rsqrtf(var + 1e-5f) * w[d] + b[d];
  out[(size_t)row * DM + d] = o;
  if (outb) outb[(size_t)row * DM + d] = f2bfu(o);
}

// out[b, p, v] = pout[(b*1031 + v)*96 + p]  for v < 1024
__global__ void final_out(const float* __restrict__ P, float* __restrict__ out,
                          int total)
{
  for (int i = blockIdx.x * blockDim.x + threadIdx.x; i < total;
       i += gridDim.x * blockDim.x) {
    int v = i & (NVAR - 1);
    int pb = i >> 10;
    int p = pb % PRED;
    int b = pb / PRED;
    out[i] = P[((size_t)b * LTOK + v) * PRED + p];
  }
}

extern "C" void kernel_launch(void* const* d_in, const int* in_sizes, int n_in,
                              void* d_out, int out_size, void* d_ws, size_t ws_size,
                              hipStream_t stream)
{
  const float* x_enc      = (const float*)d_in[0];
  const float* x_mark_enc = (const float*)d_in[1];
  const float* emb_w      = (const float*)d_in[4];
  const float* emb_b      = (const float*)d_in[5];
  const float* in_proj_w  = (const float*)d_in[6];
  const float* conv_w     = (const float*)d_in[7];
  const float* conv_b     = (const float*)d_in[8];
  const float* x_proj_w   = (const float*)d_in[9];
  const float* dt_proj_w  = (const float*)d_in[10];
  const float* dt_proj_b  = (const float*)d_in[11];
  const float* A_log      = (const float*)d_in[12];
  const float* D_param    = (const float*)d_in[13];
  const float* out_proj_w = (const float*)d_in[14];
  const float* norm1_w    = (const float*)d_in[15];
  const float* norm1_b    = (const float*)d_in[16];
  const float* norm2_w    = (const float*)d_in[17];
  const float* norm2_b    = (const float*)d_in[18];
  const float* ffn_w1     = (const float*)d_in[19];
  const float* ffn_b1     = (const float*)d_in[20];
  const float* ffn_w2     = (const float*)d_in[21];
  const float* ffn_b2     = (const float*)d_in[22];
  const float* fnorm_w    = (const float*)d_in[23];
  const float* fnorm_b    = (const float*)d_in[24];
  const float* proj_w     = (const float*)d_in[25];
  const float* proj_b     = (const float*)d_in[26];

  // ---- workspace carving (64B-aligned regions) ----------------------------
  char* base = (char*)d_ws;
  auto alloc_f = [&](size_t n) {
    float* p = (float*)base; base += ((n * 4 + 63) & ~(size_t)63); return p; };
  auto alloc_b = [&](size_t n) {
    u16* p = (u16*)base;   base += ((n * 2 + 63) & ~(size_t)63); return p; };

  float* h     = alloc_f((size_t)MPAD * DM);
  float* xz    = alloc_f((size_t)MPAD * 256);
  float* xcb   = alloc_f((size_t)MPAD * DM);
  float* dbc   = alloc_f((size_t)MPAD * DBC_LD);
  float* delta = alloc_f((size_t)MPAD * DM);
  float* t1    = xz;      // reuse (free outside the direction loop)
  float* t2    = xcb;     // reuse
  float* pout  = delta;   // reuse (MPAD x 96 <= MPAD x 128)

  u16* tokb  = alloc_b((size_t)MPAD * SEQ);
  u16* hinb  = alloc_b((size_t)MPAD * DM);
  u16* hb    = alloc_b((size_t)MPAD * DM);
  u16* xcbb  = alloc_b((size_t)MPAD * DM);
  u16* dtpb  = alloc_b((size_t)MPAD * 32);
  u16* ygb   = alloc_b((size_t)MPAD * DM);
  u16* t1b   = alloc_b((size_t)MPAD * DM);
  u16* embwb = alloc_b((size_t)DM * SEQ);
  u16* ipwb  = alloc_b((size_t)4 * 256 * DM);
  u16* xpwb  = alloc_b((size_t)4 * 48 * DM);
  u16* dpwb  = alloc_b((size_t)4 * DM * 32);
  u16* opwb  = alloc_b((size_t)4 * DM * DM);
  u16* f1wb  = alloc_b((size_t)2 * DM * DM);
  u16* f2wb  = alloc_b((size_t)2 * DM * DM);
  u16* pjwb  = alloc_b((size_t)PRED * DM);

  const dim3 W32(32);
#define GEMM(NT, ACT, ACC, Xb, LDA, Wp, BIAS, Cp, LDC, N, K)                   \
  gemm_wmma<NT, ACT, ACC><<<dim3(MPAD / 16, (N) / (16 * (NT))), W32, 0,        \
                            stream>>>((Xb), (LDA), (Wp), (BIAS), (Cp), (LDC),  \
                                      (K))
#define CVT(SRC, DST, N) to_bf16<<<256, 256, 0, stream>>>((SRC), (DST), (N))

  // ---- one-time weight conversion / padding -------------------------------
  CVT(emb_w, embwb, DM * SEQ);
  CVT(in_proj_w, ipwb, 4 * 256 * DM);
  CVT(out_proj_w, opwb, 4 * DM * DM);
  CVT(ffn_w1, f1wb, 2 * DM * DM);
  CVT(ffn_w2, f2wb, 2 * DM * DM);
  CVT(proj_w, pjwb, PRED * DM);
  pack_weights<<<64, 256, 0, stream>>>(x_proj_w, dt_proj_w, xpwb, dpwb);

  // ---- embedding: h = tokens @ emb_w^T + emb_b ----------------------------
  prep_tokens<<<2048, 256, 0, stream>>>(x_enc, x_mark_enc, tokb, MPAD * SEQ);
  GEMM(4, ACT_NONE, 0, tokb, SEQ, embwb, emb_b, h, DM, DM, SEQ);

  for (int l = 0; l < 2; ++l) {
    // freeze layer input (bf16): both direction branches read hinb
    CVT(h, hinb, MPAD * DM);
    for (int dir = 0; dir < 2; ++dir) {
      const int idx = l * 2 + dir;
      const u16*   ipw = ipwb + (size_t)idx * 256 * DM;
      const float* cw  = conv_w    + (size_t)idx * DM * 2;
      const float* cb  = conv_b    + (size_t)idx * DM;
      const u16*   xpw = xpwb + (size_t)idx * 48 * DM;
      const u16*   dpw = dpwb + (size_t)idx * DM * 32;
      const float* dpb = dt_proj_b + (size_t)idx * DM;
      const float* Al  = A_log     + (size_t)idx * DM * DSTATE;
      const float* Dp  = D_param   + (size_t)idx * DM;
      const u16*   opw = opwb + (size_t)idx * DM * DM;

      GEMM(4, ACT_NONE, 0, hinb, DM, ipw, nullptr, xz, 256, 256, DM); // in_proj
      conv_silu<<<1024, 256, 0, stream>>>(xz, cw, cb, xcb, xcbb, dir,
                                          MROWS * DM);
      GEMM(3, ACT_NONE, 0, xcbb, DM, xpw, nullptr, dbc, DBC_LD, 48, DM);
      pack_dt<<<512, 256, 0, stream>>>(dbc, dtpb, MPAD * 32);
      GEMM(4, ACT_SOFTPLUS, 0, dtpb, 32, dpw, dpb, delta, DM, DM, 32);
      mamba_scan<<<BATCH, 128, 0, stream>>>(dbc, delta, xcb, xz, Al, Dp, ygb,
                                            dir);
      GEMM(4, ACT_NONE, 1, ygb, DM, opw, nullptr, h, DM, DM, DM); // h += out
    }
    layernorm_k<<<MROWS, 128, 0, stream>>>(h, hb, h, nullptr,
                                           norm1_w + l * DM, norm1_b + l * DM);
    GEMM(4, ACT_RELU, 0, hb, DM, f1wb + (size_t)l * DM * DM, ffn_b1 + l * DM,
         t1, DM, DM, DM);
    CVT(t1, t1b, MPAD * DM);
    GEMM(4, ACT_NONE, 0, t1b, DM, f2wb + (size_t)l * DM * DM, ffn_b2 + l * DM,
         t2, DM, DM, DM);
    layernorm_k<<<MROWS, 128, 0, stream>>>(h, nullptr, h, t2,
                                           norm2_w + l * DM, norm2_b + l * DM);
  }

  layernorm_k<<<MROWS, 128, 0, stream>>>(h, hb, h, nullptr, fnorm_w, fnorm_b);
  GEMM(3, ACT_NONE, 0, hb, DM, pjwb, proj_b, pout, PRED, PRED, DM);
  final_out<<<1024, 256, 0, stream>>>(pout, (float*)d_out, BATCH * PRED * NVAR);
#undef GEMM
#undef CVT
}
